// HypergraphFusionLayer_78700980731998
// MI455X (gfx1250) — matrix operand order repaired and checked by reading, not compiled, not run on hardware
//
#include <hip/hip_runtime.h>

// ---------------------------------------------------------------------------
// HypergraphFusionLayer on MI455X (gfx1250, wave32, WMMA bf16 16x16x32)
//
// out = relu( Dv^-1/2 H De^-1 H^T Dv^-1/2 (emb @ W) )
// N=16384 nodes, E=4096 edges, F=256 features. Memory-bound: H (256MB f32)
// streamed exactly 3x (degrees, H^T-GEMM, H-GEMM) => ~768MB => ~33us floor
// @ 23.3 TB/s. GEMMs use v_wmma_f32_16x16x32_bf16 with in-register f32->bf16
// conversion; small operands (XsT 8MB, sT 2MB) stay L2-resident.
// __launch_bounds__(256,1) gives each wave a full VGPR budget so the 16 v8f
// accumulators (128 VGPRs) stay in registers (round 1 showed scratch spills).
// ---------------------------------------------------------------------------

typedef __attribute__((ext_vector_type(16))) __bf16 v16bf;
typedef __attribute__((ext_vector_type(8)))  __bf16 v8bf;
typedef __attribute__((ext_vector_type(8)))  float  v8f;
typedef __attribute__((ext_vector_type(4)))  float  v4f;

#define DEVFN __device__ __forceinline__

constexpr int   N_NODES = 16384;
constexpr int   N_EDGES = 4096;
constexpr int   F_DIM   = 256;
constexpr float EPS_F   = 1e-6f;
constexpr int   SPLITK  = 4;
constexpr int   KCHUNK  = N_NODES / SPLITK;   // 4096

// Workspace layout (bytes)
constexpr size_t OFF_DVPART = 0;                            // [16][N]   f32  1MB
constexpr size_t OFF_DEPART = 1u << 20;                     // [64][E]   f32  1MB
constexpr size_t OFF_DVIS   = 2u << 20;                     // [N]       f32  64KB
constexpr size_t OFF_DEINV  = (2u << 20) + (64u << 10);     // [E]       f32  16KB
constexpr size_t OFF_WT     = (2u << 20) + (80u << 10);     // [F][F]    f32  256KB
constexpr size_t OFF_XST    = 4u << 20;                     // [F][N]    bf16 8MB
constexpr size_t OFF_SPART  = 12u << 20;                    // [SK][E][F]f32  16MB
constexpr size_t OFF_ST     = 28u << 20;                    // [F][E]    bf16 2MB
// total ~30MB

// --- fragment helpers (wave32 WMMA 16x16x32 bf16 layouts) -------------------
// A (16x32, MxK): lane&15 = M row; lane<16: K {k..k+15},
//                 lane>=16: K {k+16..k+31}
// B (32x16, KxN): lane&15 = N col; lane<16: K {k..k+15}; lane>=16: K {k+16..k+31}
// C/D (16x16 f32): elem i -> M = i + (lane>>4)*8, N = lane&15

DEVFN v16bf load_a_f32(const float* __restrict__ lo, const float* __restrict__ hi) {
  v4f x0 = *(const v4f*)(lo);
  v4f x1 = *(const v4f*)(lo + 4);
  v4f x2 = *(const v4f*)(hi);
  v4f x3 = *(const v4f*)(hi + 4);
  v16bf a;
#pragma unroll
  for (int j = 0; j < 4; ++j) {
    a[j]      = (__bf16)x0[j];
    a[4 + j]  = (__bf16)x1[j];
    a[8 + j]  = (__bf16)x2[j];
    a[12 + j] = (__bf16)x3[j];
  }
  return a;
}

DEVFN v16bf cvt16_f32(const float* __restrict__ p) {
  v16bf b;
#pragma unroll
  for (int j = 0; j < 16; j += 4) {
    v4f x = *(const v4f*)(p + j);
    b[j]     = (__bf16)x[0];
    b[j + 1] = (__bf16)x[1];
    b[j + 2] = (__bf16)x[2];
    b[j + 3] = (__bf16)x[3];
  }
  return b;
}

DEVFN v16bf load16_bf(const __bf16* __restrict__ p) {
  union { v16bf v; v8bf h[2]; } u;
  u.h[0] = *(const v8bf*)(p);
  u.h[1] = *(const v8bf*)(p + 8);
  return u.v;
}

DEVFN v8f wmma_bf16(v16bf a, v16bf b, v8f c) {
  return __builtin_amdgcn_wmma_f32_16x16x32_bf16(false, a, false, b, (short)0, c,
                                                 false, false);
}

// --- K1: fused degree pass (one stream over H), deterministic ---------------
__global__ void deg_kernel(const float* __restrict__ H,
                           float* __restrict__ dv_part,   // [16][N]
                           float* __restrict__ de_part) { // [64][E]
  __shared__ float wavepart[256][8];
  const int t    = threadIdx.x;
  const int lane = t & 31;
  const int wid  = t >> 5;
  const int c    = blockIdx.x * 256 + t;   // edge column
  const int r0   = blockIdx.y * 256;       // node row base
  float cs = 0.f;
  for (int r = 0; r < 256; ++r) {
    float v = H[(r0 + r) * N_EDGES + c];
    cs += v;
    float ws = v;
#pragma unroll
    for (int off = 16; off >= 1; off >>= 1) ws += __shfl_xor(ws, off, 32);
    if (lane == 0) wavepart[r][wid] = ws;
  }
  de_part[blockIdx.y * N_EDGES + c] = cs;
  __syncthreads();
  float rs = 0.f;
#pragma unroll
  for (int w = 0; w < 8; ++w) rs += wavepart[t][w];
  dv_part[blockIdx.x * N_NODES + r0 + t] = rs;
}

__global__ void finalize_deg(const float* __restrict__ dv_part,
                             const float* __restrict__ de_part,
                             float* __restrict__ dvis,
                             float* __restrict__ de_inv) {
  const int i = blockIdx.x * 256 + threadIdx.x;
  float s = 0.f;
#pragma unroll
  for (int c = 0; c < 16; ++c) s += dv_part[c * N_NODES + i];
  dvis[i] = 1.0f / sqrtf(s + EPS_F);
  if (i < N_EDGES) {
    float d = 0.f;
#pragma unroll
    for (int c = 0; c < 64; ++c) d += de_part[c * N_EDGES + i];
    de_inv[i] = 1.0f / (d + EPS_F);
  }
}

// --- weight transpose so GEMM1 A-fragments are contiguous -------------------
__global__ void wt_transpose(const float* __restrict__ w, float* __restrict__ wT) {
  wT[blockIdx.x * F_DIM + threadIdx.x] = w[threadIdx.x * F_DIM + blockIdx.x];
}

// --- GEMM1: XsT[f][n] = bf16( dvis[n] * (emb @ W)[n][f] ) -------------------
// C[M=f, N=n]; A = W^T (wT rows contiguous), B = emb rows (contiguous).
// Block = 8 waves, wave owns 16 node-cols x all 256 f (16 M-tiles).
__global__ void __launch_bounds__(256, 1)
gemm1_xst(const float* __restrict__ emb,
          const float* __restrict__ wT,
          const float* __restrict__ dvis,
          __bf16* __restrict__ XsT) {
  const int lane = threadIdx.x & 31, wid = threadIdx.x >> 5;
  const int half = lane >> 4, l15 = lane & 15;
  const int n0 = blockIdx.x * 128 + wid * 16;
  const int n  = n0 + l15;
  v8f acc[16];
  const v8f zero = {0.f, 0.f, 0.f, 0.f, 0.f, 0.f, 0.f, 0.f};
#pragma unroll
  for (int m = 0; m < 16; ++m) acc[m] = zero;

#pragma unroll 1
  for (int kk = 0; kk < F_DIM; kk += 32) {
    v16bf b = cvt16_f32(emb + n * F_DIM + kk + half * 16);
#pragma unroll
    for (int mt = 0; mt < 16; ++mt) {
      const float* pa = wT + (mt * 16 + l15) * F_DIM + kk + half * 16;
      v16bf a = load_a_f32(pa, pa + 8);
      acc[mt] = wmma_bf16(a, b, acc[mt]);
    }
  }
  const float dvn = dvis[n];
#pragma unroll
  for (int mt = 0; mt < 16; ++mt)
#pragma unroll
    for (int i = 0; i < 8; ++i) {
      const int f = mt * 16 + i + half * 8;
      XsT[f * N_NODES + n] = (__bf16)(acc[mt][i] * dvn);
    }
}

// --- GEMM2: s_part[kc][e][f] = sum_{n in chunk} H[n][e] * XsT[f][n] ---------
// C[M=e, N=f]; A = H^T (strided scalar loads, coalesced across e-lanes),
// B = XsT rows (contiguous bf16). Waves own distinct e-tiles -> H read once.
__global__ void __launch_bounds__(256, 1)
gemm2_spart(const float* __restrict__ H,
            const __bf16* __restrict__ XsT,
            float* __restrict__ s_part) {
  const int lane = threadIdx.x & 31, wid = threadIdx.x >> 5;
  const int half = lane >> 4, l15 = lane & 15;
  const int e0 = blockIdx.x * 128 + wid * 16;
  const int e  = e0 + l15;
  const int kb = blockIdx.y * KCHUNK;
  v8f acc[16];
  const v8f zero = {0.f, 0.f, 0.f, 0.f, 0.f, 0.f, 0.f, 0.f};
#pragma unroll
  for (int m = 0; m < 16; ++m) acc[m] = zero;

#pragma unroll 1
  for (int kk = kb; kk < kb + KCHUNK; kk += 32) {
    const int kA = kk + half * 16;
    v16bf a;
#pragma unroll
    for (int j = 0; j < 8; ++j) {
      a[j]     = (__bf16)H[(kA + j) * N_EDGES + e];
      a[8 + j] = (__bf16)H[(kA + 8 + j) * N_EDGES + e];
    }
#pragma unroll
    for (int ft = 0; ft < 16; ++ft) {
      v16bf b = load16_bf(XsT + (ft * 16 + l15) * N_NODES + kk + half * 16);
      acc[ft] = wmma_bf16(a, b, acc[ft]);
    }
  }
  float* outp = s_part + (size_t)blockIdx.y * (N_EDGES * F_DIM);
#pragma unroll
  for (int ft = 0; ft < 16; ++ft)
#pragma unroll
    for (int i = 0; i < 8; ++i) {
      const int ee = e0 + i + half * 8;
      outp[ee * F_DIM + ft * 16 + l15] = acc[ft][i];
    }
}

// --- reduce split-K partials, scale by de^-1, transpose -> sT[f][e] bf16 ----
__global__ void reduce_scale_T(const float* __restrict__ s_part,
                               const float* __restrict__ de_inv,
                               __bf16* __restrict__ sT) {
  __shared__ float tile[32][33];
  const int t  = threadIdx.x;
  const int e0 = blockIdx.x * 32, f0 = blockIdx.y * 32;
#pragma unroll
  for (int p = 0; p < 4; ++p) {
    const int el = (t >> 5) + p * 8, fl = t & 31;
    float s = 0.f;
#pragma unroll
    for (int c = 0; c < SPLITK; ++c)
      s += s_part[(size_t)c * (N_EDGES * F_DIM) + (e0 + el) * F_DIM + f0 + fl];
    tile[el][fl] = s;
  }
  __syncthreads();
#pragma unroll
  for (int p = 0; p < 4; ++p) {
    const int fl = (t >> 5) + p * 8, el = t & 31;
    sT[(f0 + fl) * N_EDGES + e0 + el] = (__bf16)(tile[el][fl] * de_inv[e0 + el]);
  }
}

// --- GEMM3: out[n][f] = relu( dvis[n] * sum_e H[n][e] * s[e][f] ) -----------
// C[M=n, N=f]; A = H rows (contiguous b128 loads), B = sT rows (contiguous).
// Waves own distinct node-tiles -> H read once; sT (2MB) is L2-resident.
__global__ void __launch_bounds__(256, 1)
gemm3_out(const float* __restrict__ H,
          const __bf16* __restrict__ sT,
          const float* __restrict__ dvis,
          float* __restrict__ out) {
  const int lane = threadIdx.x & 31, wid = threadIdx.x >> 5;
  const int half = lane >> 4, l15 = lane & 15;
  const int n0 = blockIdx.x * 128 + wid * 16;
  const int n  = n0 + l15;
  v8f acc[16];
  const v8f zero = {0.f, 0.f, 0.f, 0.f, 0.f, 0.f, 0.f, 0.f};
#pragma unroll
  for (int m = 0; m < 16; ++m) acc[m] = zero;

#pragma unroll 1
  for (int kk = 0; kk < N_EDGES; kk += 32) {
    const float* pa = H + n * N_EDGES + kk + half * 16;
    v16bf a = load_a_f32(pa, pa + 8);
#pragma unroll
    for (int ft = 0; ft < 16; ++ft) {
      v16bf b = load16_bf(sT + (ft * 16 + l15) * N_EDGES + kk + half * 16);
      acc[ft] = wmma_bf16(a, b, acc[ft]);
    }
  }
  float dvn[8];
#pragma unroll
  for (int i = 0; i < 8; ++i) dvn[i] = dvis[n0 + i + half * 8];
#pragma unroll
  for (int ft = 0; ft < 16; ++ft)
#pragma unroll
    for (int i = 0; i < 8; ++i) {
      const int nn = n0 + i + half * 8;
      out[nn * F_DIM + ft * 16 + l15] = fmaxf(acc[ft][i] * dvn[i], 0.f);
    }
}

// ---------------------------------------------------------------------------
extern "C" void kernel_launch(void* const* d_in, const int* in_sizes, int n_in,
                              void* d_out, int out_size, void* d_ws, size_t ws_size,
                              hipStream_t stream) {
  const float* emb = (const float*)d_in[0];   // [N][256] f32
  const float* H   = (const float*)d_in[1];   // [N][E]   f32
  const float* w   = (const float*)d_in[2];   // [256][256] f32
  float* out       = (float*)d_out;           // [N][256] f32

  char* ws = (char*)d_ws;
  float*  dv_part = (float*)(ws + OFF_DVPART);
  float*  de_part = (float*)(ws + OFF_DEPART);
  float*  dvis    = (float*)(ws + OFF_DVIS);
  float*  de_inv  = (float*)(ws + OFF_DEINV);
  float*  wT      = (float*)(ws + OFF_WT);
  __bf16* XsT     = (__bf16*)(ws + OFF_XST);
  float*  s_part  = (float*)(ws + OFF_SPART);
  __bf16* sT      = (__bf16*)(ws + OFF_ST);

  deg_kernel<<<dim3(N_EDGES / 256, N_NODES / 256), 256, 0, stream>>>(H, dv_part, de_part);
  finalize_deg<<<N_NODES / 256, 256, 0, stream>>>(dv_part, de_part, dvis, de_inv);
  wt_transpose<<<F_DIM, F_DIM, 0, stream>>>(w, wT);
  gemm1_xst<<<N_NODES / 128, 256, 0, stream>>>(emb, wT, dvis, XsT);
  gemm2_spart<<<dim3(N_EDGES / 128, SPLITK), 256, 0, stream>>>(H, XsT, s_part);
  reduce_scale_T<<<dim3(N_EDGES / 32, F_DIM / 32), 256, 0, stream>>>(s_part, de_inv, sT);
  gemm3_out<<<N_NODES / 128, 256, 0, stream>>>(H, sT, dvis, out);
}
